// GLORIALocalContrastiveLoss_24910810317419
// MI455X (gfx1250) — matrix operand order repaired and checked by reading, not compiled, not run on hardware
//
#include <hip/hip_runtime.h>

// Problem constants
#define B_   48
#define D_   768
#define S_   361       // IH*IW = 19*19
#define T_   97
#define SP   384       // S padded: GEMM1 M-dim (12*32), GEMM2 K-dim, LDS row stride
#define TP   128       // T padded (4*32)
#define TEMP1_ 4.0f
#define TEMP2_ 5.0f
#define TEMP3_ 10.0f
#define EPS_   1e-8f

typedef __attribute__((ext_vector_type(16))) __bf16 v16bf;
typedef __attribute__((ext_vector_type(8)))  float  v8f;

union ABf {
  v16bf v;
  uint4 q[2];
  unsigned short u[16];
};

__device__ __forceinline__ unsigned short f2bf(float f) {
  unsigned u = __float_as_uint(f);
  u += 0x7fffu + ((u >> 16) & 1u);   // round-to-nearest-even
  return (unsigned short)(u >> 16);
}
__device__ __forceinline__ float bf2f(unsigned short h) {
  return __uint_as_float(((unsigned)h) << 16);
}
__device__ __forceinline__ v8f wmma_bf16(const ABf& a, const ABf& b, v8f c) {
  return __builtin_amdgcn_wmma_f32_16x16x32_bf16(false, a.v, false, b.v,
                                                 (short)0, c, false, false);
}

// ---------------------------------------------------------------------------
// Prep 1: img_features (B,D,S) f32 -> ctx_bf (B,D,SP) bf16 (row-major, s-padded)
//                              and -> ctxT_bf (B,SP,D) bf16 (transposed, zero pad rows)
// ---------------------------------------------------------------------------
__global__ void prep_ctx(const float* __restrict__ img,
                         unsigned short* __restrict__ ctx_bf,
                         unsigned short* __restrict__ ctxT_bf) {
  long idx = (long)blockIdx.x * blockDim.x + threadIdx.x;
  long total = (long)B_ * D_ * SP;
  if (idx >= total) return;
  int  s  = (int)(idx % SP);
  long rd = idx / SP;              // j*D + d
  int  d  = (int)(rd % D_);
  int  j  = (int)(rd / D_);
  float v = (s < S_) ? img[rd * S_ + s] : 0.0f;
  unsigned short h = f2bf(v);
  ctx_bf[rd * SP + s] = h;
  ctxT_bf[((long)j * SP + s) * D_ + d] = h;
}

// ---------------------------------------------------------------------------
// Prep 2: words_emb (B,D,T) f32 -> wordsT_bf (B,TP,D) bf16 + norm_w = ||w[:,t]||
// One block per (i,t).
// ---------------------------------------------------------------------------
__global__ void prep_words(const float* __restrict__ words,
                           unsigned short* __restrict__ wordsT_bf,
                           float* __restrict__ norm_w) {
  int bt = blockIdx.x;
  int t  = bt % TP;
  int i  = bt / TP;
  __shared__ float red[256];
  float acc = 0.f;
  for (int d = threadIdx.x; d < D_; d += 256) {
    float v = (t < T_) ? words[((long)i * D_ + d) * T_ + t] : 0.0f;
    unsigned short h = f2bf(v);
    wordsT_bf[((long)i * TP + t) * D_ + d] = h;
    float w = bf2f(h);
    acc += w * w;
  }
  red[threadIdx.x] = acc;
  __syncthreads();
  for (int off = 128; off > 0; off >>= 1) {
    if (threadIdx.x < off) red[threadIdx.x] += red[threadIdx.x + off];
    __syncthreads();
  }
  if (threadIdx.x == 0) norm_w[i * TP + t] = sqrtf(red[0]);
}

// ---------------------------------------------------------------------------
// Main fused kernel: one workgroup per (i,j) pair; 256 threads = 8 wave32.
// GEMM1 (WMMA bf16, 2x2 register blocked) -> softmax(t) -> softmax(s)
// -> GEMM2 (WMMA bf16, 2x2 blocked) fused with the cosine-sim reduction.
// LDS holds the whole (padded) 128x384 attention tile.
// ---------------------------------------------------------------------------
__global__ __launch_bounds__(256, 1)
void gloria_pair(const unsigned short* __restrict__ ctx_bf,
                 const unsigned short* __restrict__ ctxT_bf,
                 const unsigned short* __restrict__ wordsT_bf,
                 const float* __restrict__ norm_w,
                 const int* __restrict__ cap_lens,
                 float* __restrict__ sims,
                 float* __restrict__ att_out) {
  extern __shared__ float smem[];
  float* attn  = smem;                 // TP*SP   (attention tile, [t][s], stride SP)
  float* num_w = attn + TP * SP;       // 8*TP    (per-wave num accum)
  float* nc_w  = num_w + 8 * TP;       // 8*TP    (per-wave ||wei||^2 accum)
  float* red   = nc_w + 8 * TP;        // 128

  const int j    = blockIdx.x;
  const int i    = blockIdx.y;
  const int tid  = threadIdx.x;
  const int wave = tid >> 5;
  const int lane = tid & 31;
  const int nn   = lane & 15;          // N column within 16x16 subtile
  const int half = lane >> 4;          // lane half (row/K split per ISA layout)
  const int cap  = cap_lens[i];

  const unsigned short* ctxT_j = ctxT_bf   + (long)j * SP * D_;
  const unsigned short* ctx_j  = ctx_bf    + (long)j * D_ * SP;
  const unsigned short* wT_i   = wordsT_bf + (long)i * TP * D_;

  // ---------------- GEMM1: raw[s,t] = sum_d ctxT[s,d] * wordsT[t,d] ----------
  // 32x32 macro-tiles: 12 (s) x 4 (t) = 48 tiles over 8 waves.
  // A operands reused across two B tiles and vice-versa (2x arithmetic intensity).
  for (int mt = wave; mt < 12 * 4; mt += 8) {
    int s0 = (mt % 12) * 32;
    int t0 = (mt / 12) * 32;
    v8f c00 = {0.f,0.f,0.f,0.f,0.f,0.f,0.f,0.f};
    v8f c01 = c00, c10 = c00, c11 = c00;
    const uint4* pa0 = reinterpret_cast<const uint4*>(ctxT_j + (long)(s0      + nn) * D_ + 8 * half);
    const uint4* pa1 = reinterpret_cast<const uint4*>(ctxT_j + (long)(s0 + 16 + nn) * D_ + 8 * half);
    const uint4* pb0 = reinterpret_cast<const uint4*>(wT_i   + (long)(t0      + nn) * D_ + 8 * half);
    const uint4* pb1 = reinterpret_cast<const uint4*>(wT_i   + (long)(t0 + 16 + nn) * D_ + 8 * half);
    for (int k0 = 0; k0 < D_; k0 += 32) {
      ABf a0, a1, b0, b1;
      a0.q[0] = pa0[0]; a0.q[1] = pa0[2];      // K = 8h..8h+7, 16+8h..16+8h+7
      a1.q[0] = pa1[0]; a1.q[1] = pa1[2];
      b0.q[0] = pb0[0]; b0.q[1] = pb0[2];      // K = 8h..8h+7, 16+8h..16+8h+7
      b1.q[0] = pb1[0]; b1.q[1] = pb1[2];
      __builtin_prefetch((const void*)(pa0 + 16), 0, 0);   // stream-ahead hint
      pa0 += 4; pa1 += 4; pb0 += 4; pb1 += 4;
      c00 = wmma_bf16(a0, b0, c00);
      c01 = wmma_bf16(a0, b1, c01);
      c10 = wmma_bf16(a1, b0, c10);
      c11 = wmma_bf16(a1, b1, c11);
    }
    // D subtile: row M = r + 8*half (=> s), col N = nn (=> t); LDS layout [t][s].
    {
      float* d00 = &attn[(t0      + nn) * SP + s0      + 8 * half];
      float* d01 = &attn[(t0 + 16 + nn) * SP + s0      + 8 * half];
      float* d10 = &attn[(t0      + nn) * SP + s0 + 16 + 8 * half];
      float* d11 = &attn[(t0 + 16 + nn) * SP + s0 + 16 + 8 * half];
      *reinterpret_cast<float4*>(d00)     = make_float4(c00[0], c00[1], c00[2], c00[3]);
      *reinterpret_cast<float4*>(d00 + 4) = make_float4(c00[4], c00[5], c00[6], c00[7]);
      *reinterpret_cast<float4*>(d01)     = make_float4(c01[0], c01[1], c01[2], c01[3]);
      *reinterpret_cast<float4*>(d01 + 4) = make_float4(c01[4], c01[5], c01[6], c01[7]);
      *reinterpret_cast<float4*>(d10)     = make_float4(c10[0], c10[1], c10[2], c10[3]);
      *reinterpret_cast<float4*>(d10 + 4) = make_float4(c10[4], c10[5], c10[6], c10[7]);
      *reinterpret_cast<float4*>(d11)     = make_float4(c11[0], c11[1], c11[2], c11[3]);
      *reinterpret_cast<float4*>(d11 + 4) = make_float4(c11[4], c11[5], c11[6], c11[7]);
    }
  }
  __syncthreads();
  // Note: padded rows (t>=97) and padded cols (s>=361) are exact zeros because
  // ctxT/wordsT padding is zero -> GEMM1 wrote 0 there.

  // ---------------- attn1: masked softmax over t (per s) ---------------------
  for (int s = tid; s < S_; s += 256) {
    float m = -1e30f;
    for (int t = 0; t < cap; ++t) m = fmaxf(m, attn[t * SP + s]);
    float sum = 0.f;
    for (int t = 0; t < cap; ++t) {
      float e = __expf(attn[t * SP + s] - m);
      attn[t * SP + s] = e;
      sum += e;
    }
    float inv = 1.f / sum;
    for (int t = 0; t < cap; ++t) attn[t * SP + s] *= inv;
    for (int t = cap; t < T_; ++t) attn[t * SP + s] = 0.f;   // masked -> exactly 0
  }
  __syncthreads();

  // ---------------- attn2: softmax over s of attn1*TEMP1 (per t) -------------
  for (int t = tid; t < T_; t += 256) {
    float m = -1e30f;
    for (int s = 0; s < S_; ++s) m = fmaxf(m, attn[t * SP + s] * TEMP1_);
    float sum = 0.f;
    for (int s = 0; s < S_; ++s) {
      float e = __expf(attn[t * SP + s] * TEMP1_ - m);
      attn[t * SP + s] = e;
      sum += e;
    }
    float inv = 1.f / sum;
    for (int s = 0; s < S_; ++s) attn[t * SP + s] *= inv;
  }
  __syncthreads();

  // att_maps for the diagonal pair: reference wants attn2 transposed to [t][s],
  // which is exactly our LDS layout.
  if (i == j) {
    float* am = att_out + (long)i * T_ * S_;
    for (int p = tid; p < T_ * S_; p += 256) {
      int t = p / S_, s = p % S_;
      am[p] = attn[t * SP + s];
    }
  }
  for (int p = tid; p < 8 * TP * 2; p += 256) num_w[p] = 0.f;  // num_w + nc_w contiguous
  __syncthreads();

  // ---------------- GEMM2: wei[d,t] = sum_s ctx[d,s]*attn2[s,t]; fused sim ---
  // 32x32 macro-tiles: 24 (d) x 4 (t) = 96 tiles over 8 waves.
  for (int mt = wave; mt < 24 * 4; mt += 8) {
    int d0 = (mt % 24) * 32;
    int t0 = (mt / 24) * 32;
    v8f c00 = {0.f,0.f,0.f,0.f,0.f,0.f,0.f,0.f};
    v8f c01 = c00, c10 = c00, c11 = c00;
    const uint4*  pa0 = reinterpret_cast<const uint4*>(ctx_j + (long)(d0      + nn) * SP + 16 * half);
    const uint4*  pa1 = reinterpret_cast<const uint4*>(ctx_j + (long)(d0 + 16 + nn) * SP + 16 * half);
    const float4* pb0 = reinterpret_cast<const float4*>(&attn[(t0      + nn) * SP + 16 * half]);
    const float4* pb1 = reinterpret_cast<const float4*>(&attn[(t0 + 16 + nn) * SP + 16 * half]);
    for (int k0 = 0; k0 < SP; k0 += 32) {
      ABf a0, a1, b0, b1;
      a0.q[0] = pa0[0]; a0.q[1] = pa0[1];
      a1.q[0] = pa1[0]; a1.q[1] = pa1[1];
      __builtin_prefetch((const void*)(pa0 + 16), 0, 0);
      pa0 += 4; pa1 += 4;
      float4 f0 = pb0[0], f1 = pb0[1], f2 = pb0[2], f3 = pb0[3];
      float4 g0 = pb1[0], g1 = pb1[1], g2 = pb1[2], g3 = pb1[3];
      pb0 += 8; pb1 += 8;                     // advance 32 floats
      b0.u[0]  = f2bf(f0.x); b0.u[1]  = f2bf(f0.y); b0.u[2]  = f2bf(f0.z); b0.u[3]  = f2bf(f0.w);
      b0.u[4]  = f2bf(f1.x); b0.u[5]  = f2bf(f1.y); b0.u[6]  = f2bf(f1.z); b0.u[7]  = f2bf(f1.w);
      b0.u[8]  = f2bf(f2.x); b0.u[9]  = f2bf(f2.y); b0.u[10] = f2bf(f2.z); b0.u[11] = f2bf(f2.w);
      b0.u[12] = f2bf(f3.x); b0.u[13] = f2bf(f3.y); b0.u[14] = f2bf(f3.z); b0.u[15] = f2bf(f3.w);
      b1.u[0]  = f2bf(g0.x); b1.u[1]  = f2bf(g0.y); b1.u[2]  = f2bf(g0.z); b1.u[3]  = f2bf(g0.w);
      b1.u[4]  = f2bf(g1.x); b1.u[5]  = f2bf(g1.y); b1.u[6]  = f2bf(g1.z); b1.u[7]  = f2bf(g1.w);
      b1.u[8]  = f2bf(g2.x); b1.u[9]  = f2bf(g2.y); b1.u[10] = f2bf(g2.z); b1.u[11] = f2bf(g2.w);
      b1.u[12] = f2bf(g3.x); b1.u[13] = f2bf(g3.y); b1.u[14] = f2bf(g3.z); b1.u[15] = f2bf(g3.w);
      c00 = wmma_bf16(a0, b0, c00);
      c01 = wmma_bf16(a0, b1, c01);
      c10 = wmma_bf16(a1, b0, c10);
      c11 = wmma_bf16(a1, b1, c11);
    }
    // Fused epilogue per 16x16 subtile: num[t] += w[d,t]*wei[d,t], nc[t] += wei^2.
    // Lane covers d = dbase + r + 8*half for column t = tbase + nn.
    auto epi = [&](const v8f& c, int dbase, int tbase) {
      const uint4* pw = reinterpret_cast<const uint4*>(
          wT_i + (long)(tbase + nn) * D_ + dbase + 8 * half);
      ABf wv; wv.q[0] = pw[0];
      float pn = 0.f, pc = 0.f;
      for (int r = 0; r < 8; ++r) {
        float w = bf2f(wv.u[r]);
        pn += w * c[r];
        pc += c[r] * c[r];
      }
      pn += __shfl_xor(pn, 16, 32);           // combine the two d-halves
      pc += __shfl_xor(pc, 16, 32);
      if (half == 0) {                        // race-free: one lane per t per wave slot
        num_w[wave * TP + tbase + nn] += pn;
        nc_w [wave * TP + tbase + nn] += pc;
      }
    };
    epi(c00, d0,      t0);
    epi(c01, d0,      t0 + 16);
    epi(c10, d0 + 16, t0);
    epi(c11, d0 + 16, t0 + 16);
  }
  __syncthreads();

  // ---------------- sim[t] -> row scalar ------------------------------------
  for (int t = tid; t < T_; t += 256) {
    float nm = 0.f, nc = 0.f;
    for (int w = 0; w < 8; ++w) { nm += num_w[w * TP + t]; nc += nc_w[w * TP + t]; }
    float sim = nm / fmaxf(norm_w[i * TP + t] * sqrtf(nc), EPS_);
    red[t] = (t < cap) ? __expf(TEMP2_ * sim) : 0.f;
  }
  __syncthreads();
  if (tid == 0) {
    float row = 0.f;
    for (int t = 0; t < T_; ++t) row += red[t];
    // similarities[j, i] = log(row[i, j]) * TEMP3
    sims[j * B_ + i] = logf(row) * TEMP3_;
  }
}

// ---------------------------------------------------------------------------
// Final: both-direction diagonal log-softmax losses on the 48x48 matrix.
// ---------------------------------------------------------------------------
__global__ void gloria_loss(const float* __restrict__ sims, float* __restrict__ out) {
  __shared__ float c0[64], c1[64];
  int a = threadIdx.x;
  float v0 = 0.f, v1 = 0.f;
  if (a < B_) {
    float diag = sims[a * B_ + a];
    float m = -1e30f;
    for (int b = 0; b < B_; ++b) m = fmaxf(m, sims[a * B_ + b]);
    float s = 0.f;
    for (int b = 0; b < B_; ++b) s += __expf(sims[a * B_ + b] - m);
    v0 = diag - (m + logf(s));
    m = -1e30f;
    for (int b = 0; b < B_; ++b) m = fmaxf(m, sims[b * B_ + a]);
    s = 0.f;
    for (int b = 0; b < B_; ++b) s += __expf(sims[b * B_ + a] - m);
    v1 = diag - (m + logf(s));
  }
  c0[a] = v0; c1[a] = v1;
  __syncthreads();
  if (a == 0) {
    float s0 = 0.f, s1 = 0.f;
    for (int b = 0; b < B_; ++b) { s0 += c0[b]; s1 += c1[b]; }
    out[0] = -s0 / (float)B_;
    out[1] = -s1 / (float)B_;
  }
}

extern "C" void kernel_launch(void* const* d_in, const int* in_sizes, int n_in,
                              void* d_out, int out_size, void* d_ws, size_t ws_size,
                              hipStream_t stream) {
  (void)in_sizes; (void)n_in; (void)out_size; (void)ws_size;
  const float* img   = (const float*)d_in[0];   // (B, D, IH, IW) f32
  const float* words = (const float*)d_in[1];   // (B, D, T)      f32
  const int*   caps  = (const int*)d_in[2];     // (B,)           i32
  float* out = (float*)d_out;                   // [loss0, loss1, att_maps...]

  // Workspace carve-up (all 16B-aligned offsets).
  unsigned short* ctx_bf    = (unsigned short*)d_ws;                    // B*D*SP
  unsigned short* ctxT_bf   = ctx_bf   + (size_t)B_ * D_ * SP;          // B*SP*D
  unsigned short* wordsT_bf = ctxT_bf  + (size_t)B_ * SP * D_;          // B*TP*D
  float* norm_w = (float*)(wordsT_bf + (size_t)B_ * TP * D_);           // B*TP
  float* sims   = norm_w + B_ * TP;                                     // B*B

  {
    long total = (long)B_ * D_ * SP;
    int blocks = (int)((total + 255) / 256);
    prep_ctx<<<blocks, 256, 0, stream>>>(img, ctx_bf, ctxT_bf);
  }
  prep_words<<<B_ * TP, 256, 0, stream>>>(words, wordsT_bf, norm_w);

  size_t smem = (size_t)(TP * SP + 2 * 8 * TP + 128) * sizeof(float);   // ~205 KB (<320KB WGP LDS)
  (void)hipFuncSetAttribute(reinterpret_cast<const void*>(gloria_pair),
                            hipFuncAttributeMaxDynamicSharedMemorySize, (int)smem);
  dim3 grid(B_, B_);   // x = j (image), y = i (words)
  gloria_pair<<<grid, 256, smem, stream>>>(ctx_bf, ctxT_bf, wordsT_bf, norm_w,
                                           caps, sims, out + 2);
  gloria_loss<<<1, 64, 0, stream>>>(sims, out);
}
